// TrajGRU_32212254720192
// MI455X (gfx1250) — matrix operand order repaired and checked by reading, not compiled, not run on hardware
//
#include <hip/hip_runtime.h>
#include <hip/hip_bf16.h>

// ---------------------------------------------------------------------------
// TrajGRU on MI455X (gfx1250): convs as implicit GEMM on v_wmma_f32_16x16x32_f16
// Activations NHWC f16 (two 16B B-fragment loads per lane), weights device-
// packed into the exact CDNA5 A-fragment layout. FP32 accumulate, fp32 state.
// All conv geometry is constexpr -> no integer division in device code.
// Block mapping: all 8 waves of a block share one m-group (A-fragment reuse in
// WGP$) and cover 8 consecutive pixel tiles (contiguous B streams).
// ---------------------------------------------------------------------------

typedef __attribute__((ext_vector_type(8)))  _Float16 v8h;
typedef __attribute__((ext_vector_type(16))) _Float16 v16h;
typedef __attribute__((ext_vector_type(8)))  float    v8f;

union FragAB { v16h v; v8h h[2]; };

__device__ __forceinline__ int imin(int a, int b) { return a < b ? a : b; }
__device__ __forceinline__ int imax(int a, int b) { return a > b ? a : b; }

#define NEG_SLOPE 0.2f

constexpr int BN = 16, HH = 64, WWD = 64;          // batch, height, width
constexpr int NPIX = BN * HH * WWD;                // 65536 pixels

// ---------------------------------------------------------------------------
// Implicit-GEMM conv, all geometry compile-time.
//   act : NHWC f16, LDC channels/pixel; conv consumes channels [0, CIN),
//         K ordered (kh, kw, cin), cin chunked by 32.
//   wpk : f16 A-fragments: element(((kc*mtiles + mt)*32 + lane)*16 + i)
//   EPI 0: fp32 NCHW (+bias), guard co < COUTREAL
//   EPI 1: leaky(out+bias) -> f16 NHWC (ldc_out = COUTREAL)
// Wave tile: MW m-tiles x NW 16-pixel tiles (pixels share b,y; W%16*NW==0).
// Per K-chunk: NW B-frag loads + MW A-frag loads -> MW*NW v_wmma.
// ---------------------------------------------------------------------------
template <int KH, int KW, int PAD, int LDC, int CIN, int COUTP, int COUTREAL,
          int MW, int NW, int EPI>
__global__ __launch_bounds__(256) void conv_wmma(
    const _Float16* __restrict__ act, const _Float16* __restrict__ wpk,
    const float* __restrict__ bias, float* __restrict__ outF32,
    _Float16* __restrict__ outF16) {
  constexpr int mGroups   = COUTP / (16 * MW);
  constexpr int pixGroups = NPIX / (16 * NW);      // power of two
  constexpr int CPT       = CIN / 32;              // cin chunks per tap
  constexpr int mtiles    = COUTP / 16;
  constexpr int HW        = HH * WWD;

  const int lane = threadIdx.x & 31;
  const int wave = threadIdx.x >> 5;
  const int tile = blockIdx.x * 8 + wave;          // 8 waves / block
  if (tile >= mGroups * pixGroups) return;         // wave-uniform exit
  // pixGroups is POT and a multiple of 8: div/mod are shifts, and all waves of
  // a block share mg -> shared A fragments, consecutive pg -> contiguous B.
  const int mg = tile / pixGroups;
  const int pg = tile % pixGroups;
  const int n0 = pg * (16 * NW);
  const int x0 = n0 % WWD;                         // multiple of 16*NW
  const int y  = (n0 / WWD) % HH;
  const int b  = n0 / HW;
  const int laneHi = lane >> 4;                    // 0|1 : K-half / M-half
  const int xl = x0 + (lane & 15);

  v8f acc[MW][NW];
#pragma unroll
  for (int mw = 0; mw < MW; ++mw)
#pragma unroll
    for (int nw = 0; nw < NW; ++nw) { v8f z = {}; acc[mw][nw] = z; }

  for (int kh = 0; kh < KH; ++kh) {
    const int iy = y + kh - PAD;
    const bool rowOK = (unsigned)iy < (unsigned)HH;
    for (int kw = 0; kw < KW; ++kw) {
      for (int cc = 0; cc < CPT; ++cc) {
        // --- B fragments: two 16-byte loads per lane (channels +0, +16) from NHWC ---
        FragAB bfr[NW];
#pragma unroll
        for (int nw = 0; nw < NW; ++nw) {
          const int ix = xl + nw * 16 + kw - PAD;
          if (rowOK && (unsigned)ix < (unsigned)WWD) {
            const _Float16* p = act +
                (size_t)((b * HH + iy) * WWD + ix) * LDC + (cc << 5) +
                laneHi * 8;
            bfr[nw].h[0] = *(const v8h*)(p);
            bfr[nw].h[1] = *(const v8h*)(p + 16);
          } else {
            v16h z = {}; bfr[nw].v = z;            // zeros padding
          }
        }
        // --- A fragments + WMMA ---
        const int kc = (kh * KW + kw) * CPT + cc;
        const _Float16* wp = wpk + (size_t)kc * mtiles * 512 + lane * 16;
        __builtin_prefetch((const void*)(wp + (size_t)mtiles * 512), 0, 3);
#pragma unroll
        for (int mw = 0; mw < MW; ++mw) {
          FragAB afr;
          const _Float16* ap = wp + (size_t)(mg * MW + mw) * 512;
          afr.h[0] = *(const v8h*)(ap);
          afr.h[1] = *(const v8h*)(ap + 8);
#pragma unroll
          for (int nw = 0; nw < NW; ++nw)
            acc[mw][nw] = __builtin_amdgcn_wmma_f32_16x16x32_f16(
                false, afr.v, false, bfr[nw].v, (short)0, acc[mw][nw], false,
                false);
        }
      }
    }
  }

  // --- epilogue ---
#pragma unroll
  for (int mw = 0; mw < MW; ++mw) {
    const int coBase = (mg * MW + mw) * 16 + laneHi * 8;  // D: M = r + hi*8
#pragma unroll
    for (int nw = 0; nw < NW; ++nw) {
      const int x = xl + nw * 16;
      if (EPI == 0) {
        float* o = outF32 + ((size_t)b * COUTREAL + coBase) * HW + y * WWD + x;
#pragma unroll
        for (int r = 0; r < 8; ++r) {
          const int co = coBase + r;
          if (co < COUTREAL) o[(size_t)r * HW] = acc[mw][nw][r] + bias[co];
        }
      } else {
        alignas(16) _Float16 tmp[8];
#pragma unroll
        for (int r = 0; r < 8; ++r) {
          float v = acc[mw][nw][r] + bias[coBase + r];
          v = v >= 0.f ? v : NEG_SLOPE * v;
          tmp[r] = (_Float16)v;
        }
        *(v8h*)(outF16 + (size_t)((b * HH + y) * WWD + x) * COUTREAL + coBase) =
            *(const v8h*)tmp;
      }
    }
  }
}

// ---------------------------------------------------------------------------
// Pack OIHW fp32 weights -> f16 A-fragments.
// A 16x32 layout: lane<16: K {i<8: i, i>=8: i+8}; lane>=16: +8. M = lane&15.
// K ordered (kh, kw, cin).
// ---------------------------------------------------------------------------
__global__ void pack_weights(const float* __restrict__ w, _Float16* __restrict__ out,
                             int cout, int coutP, int cin, int KH, int KW) {
  const int idx = blockIdx.x * 256 + threadIdx.x;
  const int mtiles = coutP >> 4;
  const int total = (cin * KH * KW / 32) * mtiles * 512;
  if (idx >= total) return;
  const int i    = idx & 15;
  const int lane = (idx >> 4) & 31;
  const int mt   = (idx >> 9) % mtiles;
  const int kc   = idx / (512 * mtiles);
  const int m    = mt * 16 + (lane & 15);
  const int koff = (i < 8 ? i : i + 8) + ((lane >= 16) ? 8 : 0);
  const int K    = kc * 32 + koff;
  const int ci   = K % cin;
  const int tap  = K / cin;
  const int kh = tap / KW, kw = tap % KW;
  float v = 0.f;
  if (m < cout) v = w[(((size_t)m * cin + ci) * KH + kh) * KW + kw];
  out[idx] = (_Float16)v;
}

// Merge [32,64,5,5] + [32,64,5,5] -> [32,128,5,5] (conv over concat(x,h)).
__global__ void merge_wcat(const float* __restrict__ wa, const float* __restrict__ wb,
                           float* __restrict__ o) {
  const int idx = blockIdx.x * 256 + threadIdx.x;
  if (idx >= 32 * 128 * 25) return;
  const int t = idx % 25;
  const int c = (idx / 25) % 128;
  const int oc = idx / (25 * 128);
  o[idx] = (c < 64) ? wa[((size_t)oc * 64 + c) * 25 + t]
                    : wb[((size_t)oc * 64 + (c - 64)) * 25 + t];
}

__global__ void bias_sum(const float* a, const float* b, float* o, int n) {
  const int i = blockIdx.x * 256 + threadIdx.x;
  if (i < n) o[i] = a[i] + b[i];
}

// Build NHWC f16 concat(x_t, prev_h); prev_h==nullptr -> zeros (t==0).
__global__ void pack_concat(const float* __restrict__ x, const float* __restrict__ h,
                            _Float16* __restrict__ cat) {
  const int idx = blockIdx.x * 256 + threadIdx.x;   // (pixel, 8-ch group)
  if (idx >= NPIX * 16) return;
  const int g = idx & 15;
  const int p = idx >> 4;
  const int xw = p % WWD, yw = (p / WWD) % HH, bw = p / (HH * WWD);
  constexpr size_t HW = (size_t)HH * WWD;
  const int c0 = g * 8;
  alignas(16) _Float16 tmp[8];
  if (c0 < 64) {
    const float* s = x + ((size_t)bw * 64 + c0) * HW + (size_t)yw * WWD + xw;
#pragma unroll
    for (int r = 0; r < 8; ++r) tmp[r] = (_Float16)s[(size_t)r * HW];
  } else if (h) {
    const float* s = h + ((size_t)bw * 64 + (c0 - 64)) * HW + (size_t)yw * WWD + xw;
#pragma unroll
    for (int r = 0; r < 8; ++r) tmp[r] = (_Float16)s[(size_t)r * HW];
  } else {
#pragma unroll
    for (int r = 0; r < 8; ++r) tmp[r] = (_Float16)0.f;
  }
  *(v8h*)(cat + (size_t)p * 128 + c0) = *(const v8h*)tmp;
}

// Bilinear warp (grid_sample, zeros padding, pixel coords) -> f16 NHWC [B,HW,320].
__global__ void warp_bilinear(const float* __restrict__ flows,  // [B,10,H,W]
                              const float* __restrict__ h,      // [B,64,H,W] | null
                              _Float16* __restrict__ warped) {
  const int idx = blockIdx.x * 256 + threadIdx.x;   // (b, l, y, x)
  if (idx >= BN * 5 * HH * WWD) return;
  const int xw = idx % WWD, yw = (idx / WWD) % HH;
  const int l = (idx / (WWD * HH)) % 5, bw = idx / (WWD * HH * 5);
  constexpr size_t HW = (size_t)HH * WWD;
  const size_t sp = (size_t)yw * WWD + xw;
  _Float16* o = warped + (size_t)((bw * HH + yw) * WWD + xw) * 320 + l * 64;
  if (!h) {
    v8h z = {};
#pragma unroll
    for (int g = 0; g < 8; ++g) ((v8h*)o)[g] = z;
    return;
  }
  const float px = (float)xw - flows[((size_t)bw * 10 + l * 2 + 0) * HW + sp];
  const float py = (float)yw - flows[((size_t)bw * 10 + l * 2 + 1) * HW + sp];
  const float x0f = floorf(px), y0f = floorf(py);
  const float wx = px - x0f, wy = py - y0f;
  const int x0 = (int)x0f, y0 = (int)y0f, x1 = x0 + 1, y1 = y0 + 1;
  const float v00 = (x0 >= 0 && x0 < WWD && y0 >= 0 && y0 < HH) ? 1.f : 0.f;
  const float v01 = (x1 >= 0 && x1 < WWD && y0 >= 0 && y0 < HH) ? 1.f : 0.f;
  const float v10 = (x0 >= 0 && x0 < WWD && y1 >= 0 && y1 < HH) ? 1.f : 0.f;
  const float v11 = (x1 >= 0 && x1 < WWD && y1 >= 0 && y1 < HH) ? 1.f : 0.f;
  const int x0c = imin(imax(x0, 0), WWD - 1), x1c = imin(imax(x1, 0), WWD - 1);
  const int y0c = imin(imax(y0, 0), HH - 1), y1c = imin(imax(y1, 0), HH - 1);
  const size_t i00 = (size_t)y0c * WWD + x0c, i01 = (size_t)y0c * WWD + x1c;
  const size_t i10 = (size_t)y1c * WWD + x0c, i11 = (size_t)y1c * WWD + x1c;
  const float w00 = (1.f - wx) * (1.f - wy) * v00, w01 = wx * (1.f - wy) * v01;
  const float w10 = (1.f - wx) * wy * v10, w11 = wx * wy * v11;
  const float* hb = h + (size_t)bw * 64 * HW;
  for (int c = 0; c < 64; ++c) {
    const float* hc = hb + (size_t)c * HW;
    o[c] = (_Float16)(hc[i00] * w00 + hc[i01] * w01 + hc[i10] * w10 + hc[i11] * w11);
  }
}

// GRU gate: r/u = sigmoid, m = leaky, h' = u*h + (1-u)*m.
__global__ void gru_gate(const float* __restrict__ i2h, const float* __restrict__ h2h,
                         const float* __restrict__ hprev, float* __restrict__ hnext) {
  const int idx = blockIdx.x * 256 + threadIdx.x;
  constexpr int HW = HH * WWD;
  if (idx >= BN * 64 * HW) return;
  const int p = idx % HW, c = (idx / HW) % 64, bw = idx / (HW * 64);
  const size_t base = (size_t)bw * 192 * HW;
  const float ir = i2h[base + (size_t)c * HW + p];
  const float iu = i2h[base + (size_t)(c + 64) * HW + p];
  const float im = i2h[base + (size_t)(c + 128) * HW + p];
  const float hr = h2h[base + (size_t)c * HW + p];
  const float hu = h2h[base + (size_t)(c + 64) * HW + p];
  const float hm = h2h[base + (size_t)(c + 128) * HW + p];
  const float hp = hprev ? hprev[((size_t)bw * 64 + c) * HW + p] : 0.f;
  const float r = 1.f / (1.f + __expf(-(ir + hr)));
  const float u = 1.f / (1.f + __expf(-(iu + hu)));
  float m = im + r * hm;
  m = m >= 0.f ? m : NEG_SLOPE * m;
  hnext[((size_t)bw * 64 + c) * HW + p] = u * hp + (1.f - u) * m;
}

// ---------------------------------------------------------------------------
extern "C" void kernel_launch(void* const* d_in, const int* in_sizes, int n_in,
                              void* d_out, int out_size, void* d_ws, size_t ws_size,
                              hipStream_t stream) {
  const float* x     = (const float*)d_in[0];
  const float* wi2h  = (const float*)d_in[1];
  const float* bi2h  = (const float*)d_in[2];
  const float* wi2f  = (const float*)d_in[3];
  const float* bi2f  = (const float*)d_in[4];
  const float* wh2f  = (const float*)d_in[5];
  const float* bh2f  = (const float*)d_in[6];
  const float* wflow = (const float*)d_in[7];
  const float* bflow = (const float*)d_in[8];
  const float* wret  = (const float*)d_in[9];
  const float* bret  = (const float*)d_in[10];
  float* outp = (float*)d_out;

  constexpr int S = 10;
  constexpr size_t HW = (size_t)HH * WWD;
  constexpr size_t hElems = (size_t)BN * 64 * HW;      // one state slice

  // ---- workspace carve (256B aligned) ----
  char* wsp = (char*)d_ws;
  auto carve = [&](size_t bytes) -> char* {
    char* p = wsp;
    wsp += (bytes + 255) & ~(size_t)255;
    return p;
  };
  _Float16* cat    = (_Float16*)carve((size_t)BN * HW * 128 * 2);  // NHWC [x;h]
  _Float16* fbuf   = (_Float16*)carve((size_t)BN * HW * 32 * 2);   // NHWC f
  _Float16* warped = (_Float16*)carve((size_t)BN * HW * 320 * 2);  // NHWC warped
  float* i2ht  = (float*)carve((size_t)BN * 192 * HW * 4);
  float* flows = (float*)carve((size_t)BN * 10 * HW * 4);
  float* h2h   = (float*)carve((size_t)BN * 192 * HW * 4);
  float* wcat  = (float*)carve((size_t)32 * 128 * 25 * 4);
  float* bcat  = (float*)carve(32 * 4);
  _Float16* pw_i2h  = (_Float16*)carve((size_t)576 * 192 * 2);
  _Float16* pw_cat  = (_Float16*)carve((size_t)3200 * 32 * 2);
  _Float16* pw_flow = (_Float16*)carve((size_t)800 * 16 * 2);
  _Float16* pw_ret  = (_Float16*)carve((size_t)320 * 192 * 2);

  // ---- per-call (deterministic, cheap) weight prep ----
  merge_wcat<<<(32 * 128 * 25 + 255) / 256, 256, 0, stream>>>(wi2f, wh2f, wcat);
  bias_sum<<<1, 256, 0, stream>>>(bi2f, bh2f, bcat, 32);
  pack_weights<<<(110592 + 255) / 256, 256, 0, stream>>>(wi2h, pw_i2h, 192, 192, 64, 3, 3);
  pack_weights<<<(102400 + 255) / 256, 256, 0, stream>>>(wcat, pw_cat, 32, 32, 128, 5, 5);
  pack_weights<<<(12800 + 255) / 256, 256, 0, stream>>>(wflow, pw_flow, 10, 16, 32, 5, 5);
  pack_weights<<<(61440 + 255) / 256, 256, 0, stream>>>(wret, pw_ret, 192, 192, 320, 1, 1);

  // tiles = mGroups * pixGroups; blocks = ceil(tiles / 8 waves)
  constexpr int blk_i2h  = (3 * (NPIX / 32) + 7) / 8;   // MW=4,NW=2 -> 768
  constexpr int blk_f    = (1 * (NPIX / 32) + 7) / 8;   // MW=2,NW=2 -> 256
  constexpr int blk_flow = (1 * (NPIX / 64) + 7) / 8;   // MW=1,NW=4 -> 128
  constexpr int blk_ret  = (3 * (NPIX / 32) + 7) / 8;   // MW=4,NW=2 -> 768

  for (int t = 0; t < S; ++t) {
    const float* xt    = x + (size_t)t * BN * 64 * HW;
    const float* hprev = (t == 0) ? nullptr : outp + (size_t)(t - 1) * hElems;
    float* hnext       = outp + (size_t)t * hElems;

    // 1) concat(x_t, h) -> NHWC f16
    pack_concat<<<(NPIX * 16 + 255) / 256, 256, 0, stream>>>(xt, hprev, cat);

    // 2) i2h: 3x3, cin 64 (channels 0..63 of cat), cout 192 -> fp32 NCHW
    conv_wmma<3, 3, 1, 128, 64, 192, 192, 4, 2, 0>
        <<<blk_i2h, 256, 0, stream>>>(cat, pw_i2h, bi2h, i2ht, nullptr);

    // 3) f = leaky(conv5x5(cat)): cin 128, cout 32 -> f16 NHWC
    conv_wmma<5, 5, 2, 128, 128, 32, 32, 2, 2, 1>
        <<<blk_f, 256, 0, stream>>>(cat, pw_cat, bcat, nullptr, fbuf);

    // 4) flows = conv5x5(f): cin 32, cout 10 (padded 16) -> fp32 NCHW
    conv_wmma<5, 5, 2, 32, 32, 16, 10, 1, 4, 0>
        <<<blk_flow, 256, 0, stream>>>(fbuf, pw_flow, bflow, flows, nullptr);

    // 5) warp prev_h by -flow (bilinear, zeros) -> f16 NHWC [B,HW,320]
    warp_bilinear<<<(BN * 5 * HH * WWD + 255) / 256, 256, 0, stream>>>(
        flows, hprev, warped);

    // 6) h2h = conv1x1(warped): cin 320, cout 192 -> fp32 NCHW
    conv_wmma<1, 1, 0, 320, 320, 192, 192, 4, 2, 0>
        <<<blk_ret, 256, 0, stream>>>(warped, pw_ret, bret, h2h, nullptr);

    // 7) gates -> outs[t]
    gru_gate<<<((int)(BN * 64 * HW) + 255) / 256, 256, 0, stream>>>(
        i2ht, h2h, hprev, hnext);
  }

  // last_h = outs[S-1]
  hipMemcpyAsync(outp + (size_t)S * hElems, outp + (size_t)(S - 1) * hElems,
                 hElems * sizeof(float), hipMemcpyDeviceToDevice, stream);
}
